// GNN_GATConv_43061342109909
// MI455X (gfx1250) — matrix operand order repaired and checked by reading, not compiled, not run on hardware
//
#include <hip/hip_runtime.h>
#include <stdint.h>

typedef __attribute__((ext_vector_type(16))) __bf16 v16bf;
typedef __attribute__((ext_vector_type(8)))  float  v8f;

#define N_NODES 10000
#define N_EDGES 80000
#define N_ALL   (N_EDGES + N_NODES)

struct alignas(16) U4 { unsigned x, y, z, w; };
union FragBF { U4 u[2]; v16bf v; };

static __device__ __forceinline__ unsigned short f32_to_bf16(float f) {
  unsigned u = __float_as_uint(f);
  unsigned r = (u + 0x7FFFu + ((u >> 16) & 1u)) >> 16;   // round-to-nearest-even
  return (unsigned short)r;
}
static __device__ __forceinline__ unsigned f2ord(float f) {  // monotonic float->uint
  int i = __float_as_int(f);
  return (i >= 0) ? ((unsigned)i | 0x80000000u) : ~(unsigned)i;
}
static __device__ __forceinline__ float ord2f(unsigned o) {
  int i = (o & 0x80000000u) ? (int)(o & 0x7FFFFFFFu) : (int)~o;
  return __int_as_float(i);
}

// ---------------- conversion / transpose ----------------
__global__ void k_cvt_bf16(const float* __restrict__ src, unsigned short* __restrict__ dst, int n) {
  int i = blockIdx.x * blockDim.x + threadIdx.x;
  if (i < n) dst[i] = f32_to_bf16(src[i]);
}

// W[k,n] (row-major KxN, f32) -> Wt[n,k] (row-major NxK, bf16)
__global__ void k_transpose_bf16(const float* __restrict__ W, unsigned short* __restrict__ Wt,
                                 int K, int N) {
  int i = blockIdx.x * blockDim.x + threadIdx.x;
  if (i >= K * N) return;
  int k = i / N, n = i - k * N;
  Wt[(size_t)n * K + k] = f32_to_bf16(W[i]);
}

// ---------------- bf16 WMMA GEMM: C[M,N] = A[M,K] * Wt[N,K]^T ----------------
// One wave computes a 16x64 strip (4 WMMA tiles sharing the A fragment).
__global__ __launch_bounds__(256)
void k_gemm_bf16(const unsigned short* __restrict__ A, const unsigned short* __restrict__ Bt,
                 float* __restrict__ C, int N, int K) {
  const int M_TILES = N_NODES / 16;            // 625
  int lane = threadIdx.x & 31;
  int wid  = blockIdx.x * 8 + (threadIdx.x >> 5);
  int nt4  = N >> 6;
  if (wid >= M_TILES * nt4) return;
  int tn4  = wid / M_TILES;
  int tm   = wid - tn4 * M_TILES;
  int row0 = tm << 4;
  int col0 = tn4 << 6;
  int n16  = lane & 15;
  int half = lane >> 4;

  // A fragment (16x32 bf16): lane half selects K 0..15 / 16..31
  const unsigned short* pa  = A  + (size_t)(row0 + n16) * K + half * 16;
  // B fragment (32x16 bf16, from transposed weights): lane half selects K 0..15 / 16..31
  const unsigned short* pb0 = Bt + (size_t)(col0 + n16) * K + half * 16;
  size_t bstride = (size_t)16 * K;

  v8f acc0 = {0,0,0,0,0,0,0,0};
  v8f acc1 = acc0, acc2 = acc0, acc3 = acc0;

  for (int k0 = 0; k0 < K; k0 += 32) {
    FragBF a;
    a.u[0] = *(const U4*)(pa + k0);
    a.u[1] = *(const U4*)(pa + k0 + 8);
    v16bf b0 = *(const v16bf*)(pb0 + k0);
    v16bf b1 = *(const v16bf*)(pb0 + bstride + k0);
    v16bf b2 = *(const v16bf*)(pb0 + 2 * bstride + k0);
    v16bf b3 = *(const v16bf*)(pb0 + 3 * bstride + k0);
    acc0 = __builtin_amdgcn_wmma_f32_16x16x32_bf16(false, a.v, false, b0, (short)0, acc0, false, false);
    acc1 = __builtin_amdgcn_wmma_f32_16x16x32_bf16(false, a.v, false, b1, (short)0, acc1, false, false);
    acc2 = __builtin_amdgcn_wmma_f32_16x16x32_bf16(false, a.v, false, b2, (short)0, acc2, false, false);
    acc3 = __builtin_amdgcn_wmma_f32_16x16x32_bf16(false, a.v, false, b3, (short)0, acc3, false, false);
  }

  float* c0 = C + (size_t)(row0 + half * 8) * N + col0 + n16;
#pragma unroll
  for (int r = 0; r < 8; ++r) {
    c0[(size_t)r * N +  0] = acc0[r];
    c0[(size_t)r * N + 16] = acc1[r];
    c0[(size_t)r * N + 32] = acc2[r];
    c0[(size_t)r * N + 48] = acc3[r];
  }
}

// ---------------- attention pieces ----------------
// al_src[row] = xw[row,:].a_src ; al_dst[row] = xw[row,:].a_dst   (one wave per row)
__global__ __launch_bounds__(256)
void k_rowdot2(const float* __restrict__ XW, const float* __restrict__ avs,
               const float* __restrict__ avd, float* __restrict__ als,
               float* __restrict__ ald, int F) {
  int lane = threadIdx.x & 31;
  int row  = blockIdx.x * 8 + (threadIdx.x >> 5);
  if (row >= N_NODES) return;
  const float* p = XW + (size_t)row * F;
  float s = 0.f, d = 0.f;
  for (int f = lane; f < F; f += 32) { float v = p[f]; s += v * avs[f]; d += v * avd[f]; }
  for (int o = 16; o > 0; o >>= 1) { s += __shfl_down(s, o); d += __shfl_down(d, o); }
  if (lane == 0) { als[row] = s; ald[row] = d; }
}

__global__ void k_zero_f32(float* __restrict__ p, size_t n) {
  size_t i = (size_t)blockIdx.x * blockDim.x + threadIdx.x;
  if (i < n) p[i] = 0.f;
}
__global__ void k_init_md(unsigned* __restrict__ m, float* __restrict__ denom, int n) {
  int i = blockIdx.x * blockDim.x + threadIdx.x;
  if (i < n) { m[i] = 0x007FFFFFu /* ord(-inf) */; denom[i] = 0.f; }
}

__global__ void k_edge_logit_max(const long long* __restrict__ src, const long long* __restrict__ dst,
                                 const float* __restrict__ als, const float* __restrict__ ald,
                                 float* __restrict__ ev, unsigned* __restrict__ m) {
  int e = blockIdx.x * blockDim.x + threadIdx.x;
  if (e >= N_ALL) return;
  int s, d;
  if (e < N_EDGES) { s = (int)src[e]; d = (int)dst[e]; } else { s = d = e - N_EDGES; }
  float v = als[s] + ald[d];
  v = (v > 0.f) ? v : 0.2f * v;                 // leaky_relu(0.2)
  ev[e] = v;
  atomicMax(&m[d], f2ord(v));
}

__global__ void k_edge_exp(const long long* __restrict__ dst, const float* __restrict__ ev,
                           const unsigned* __restrict__ m, float* __restrict__ ex,
                           float* __restrict__ denom) {
  int e = blockIdx.x * blockDim.x + threadIdx.x;
  if (e >= N_ALL) return;
  int d = (e < N_EDGES) ? (int)dst[e] : e - N_EDGES;
  float x = expf(ev[e] - ord2f(m[d]));
  ex[e] = x;
  atomicAdd(&denom[d], x);
}

// out[dst,:] += alpha_e * xw[src,:]   (one 256-thread block per edge)
__global__ __launch_bounds__(256)
void k_aggregate(const long long* __restrict__ src, const long long* __restrict__ dst,
                 const float* __restrict__ ex, const float* __restrict__ denom,
                 const float* __restrict__ XW, float* __restrict__ out, int F) {
  int e = blockIdx.x;
  int s, d;
  if (e < N_EDGES) { s = (int)src[e]; d = (int)dst[e]; } else { s = d = e - N_EDGES; }
  float coef = ex[e] / (denom[d] + 1e-16f);
  const float* xs = XW + (size_t)s * F;
  float* od = out + (size_t)d * F;
  for (int f = threadIdx.x; f < F; f += 256) atomicAdd(&od[f], coef * xs[f]);
}

// ---------------- epilogues ----------------
__global__ void k_epilogue_bf16(const float* __restrict__ acc, const float* __restrict__ bias,
                                unsigned short* __restrict__ hbf, int F, int total) {
  int i = blockIdx.x * blockDim.x + threadIdx.x;
  if (i >= total) return;
  float v = acc[i] + bias[i % F];
  hbf[i] = f32_to_bf16(v > 0.f ? v : 0.f);
}
__global__ void k_epilogue_f32(const float* __restrict__ acc, const float* __restrict__ bias,
                               float* __restrict__ h, int F, int total) {
  int i = blockIdx.x * blockDim.x + threadIdx.x;
  if (i >= total) return;
  float v = acc[i] + bias[i % F];
  h[i] = v > 0.f ? v : 0.f;
}

// L2 normalize rows (one wave per row)
__global__ __launch_bounds__(256)
void k_normalize(const float* __restrict__ h, float* __restrict__ out, int F) {
  int lane = threadIdx.x & 31;
  int row  = blockIdx.x * 8 + (threadIdx.x >> 5);
  if (row >= N_NODES) return;
  const float* p = h + (size_t)row * F;
  float ss = 0.f;
  for (int f = lane; f < F; f += 32) { float v = p[f]; ss += v * v; }
  for (int o = 16; o > 0; o >>= 1) ss += __shfl_down(ss, o);
  float inv = 1.f / fmaxf(sqrtf(__shfl(ss, 0)), 1e-12f);
  float* o_ = out + (size_t)row * F;
  for (int f = lane; f < F; f += 32) o_[f] = p[f] * inv;
}

// ---------------- host-side layer driver ----------------
struct Scratch {
  unsigned short* Abf; unsigned short* Wt;
  float* XW; float* OUT;
  float* als; float* ald; unsigned* mo; float* denom; float* ev; float* ex;
};

static void run_layer(const Scratch& S, int K, int N, const float* W,
                      const float* avs, const float* avd,
                      const long long* esrc, const long long* edst, hipStream_t stream) {
  k_transpose_bf16<<<(K * N + 255) / 256, 256, 0, stream>>>(W, S.Wt, K, N);
  int tiles = (N_NODES / 16) * (N >> 6);
  k_gemm_bf16<<<(tiles + 7) / 8, 256, 0, stream>>>(S.Abf, S.Wt, S.XW, N, K);
  k_rowdot2<<<(N_NODES + 7) / 8, 256, 0, stream>>>(S.XW, avs, avd, S.als, S.ald, N);
  size_t tot = (size_t)N_NODES * N;
  k_zero_f32<<<(unsigned)((tot + 255) / 256), 256, 0, stream>>>(S.OUT, tot);
  k_init_md<<<(N_NODES + 255) / 256, 256, 0, stream>>>(S.mo, S.denom, N_NODES);
  k_edge_logit_max<<<(N_ALL + 255) / 256, 256, 0, stream>>>(esrc, edst, S.als, S.ald, S.ev, S.mo);
  k_edge_exp<<<(N_ALL + 255) / 256, 256, 0, stream>>>(edst, S.ev, S.mo, S.ex, S.denom);
  k_aggregate<<<N_ALL, 256, 0, stream>>>(esrc, edst, S.ex, S.denom, S.XW, S.OUT, N);
}

extern "C" void kernel_launch(void* const* d_in, const int* in_sizes, int n_in,
                              void* d_out, int out_size, void* d_ws, size_t ws_size,
                              hipStream_t stream) {
  (void)in_sizes; (void)n_in; (void)out_size; (void)ws_size;
  const float*     x   = (const float*)d_in[0];
  const long long* ei  = (const long long*)d_in[1];   // int64 [2,E]
  const float* W1 = (const float*)d_in[2];
  const float* W2 = (const float*)d_in[3];
  const float* W3 = (const float*)d_in[4];
  const float* a1s = (const float*)d_in[5];
  const float* a1d = (const float*)d_in[6];
  const float* a2s = (const float*)d_in[7];
  const float* a2d = (const float*)d_in[8];
  const float* a3s = (const float*)d_in[9];
  const float* a3d = (const float*)d_in[10];
  const float* b1 = (const float*)d_in[11];
  const float* b2 = (const float*)d_in[12];
  const float* b3 = (const float*)d_in[13];
  const long long* esrc = ei;
  const long long* edst = ei + N_EDGES;

  char* ws = (char*)d_ws;
  Scratch S;
  S.Abf   = (unsigned short*)(ws);                 // 10000*2048 bf16 = 40,960,000 B
  S.Wt    = (unsigned short*)(ws + 40960000);      // 2048*2048 bf16  =  8,388,608 B
  S.XW    = (float*)(ws + 49348608);               // 10000*2048 f32  = 81,920,000 B
  S.OUT   = (float*)(ws + 131268608);              // 10000*2048 f32  = 81,920,000 B
  S.als   = (float*)(ws + 213188608);
  S.ald   = (float*)(ws + 213229568);
  S.mo    = (unsigned*)(ws + 213270528);
  S.denom = (float*)(ws + 213311488);
  S.ev    = (float*)(ws + 213352448);
  S.ex    = (float*)(ws + 213712896);              // end ~214.1 MB

  // ---- layer 1: Fin=256, Fout=2048 ----
  k_cvt_bf16<<<(N_NODES * 256 + 255) / 256, 256, 0, stream>>>(x, S.Abf, N_NODES * 256);
  run_layer(S, 256, 2048, W1, a1s, a1d, esrc, edst, stream);
  k_epilogue_bf16<<<(N_NODES * 2048 + 255) / 256, 256, 0, stream>>>(S.OUT, b1, S.Abf, 2048, N_NODES * 2048);

  // ---- layer 2: Fin=2048, Fout=2048 ----
  run_layer(S, 2048, 2048, W2, a2s, a2d, esrc, edst, stream);
  k_epilogue_bf16<<<(N_NODES * 2048 + 255) / 256, 256, 0, stream>>>(S.OUT, b2, S.Abf, 2048, N_NODES * 2048);

  // ---- layer 3: Fin=2048, Fout=1024 ----
  run_layer(S, 2048, 1024, W3, a3s, a3d, esrc, edst, stream);
  k_epilogue_f32<<<(N_NODES * 1024 + 255) / 256, 256, 0, stream>>>(S.OUT, b3, S.XW, 1024, N_NODES * 1024);

  // ---- final row-wise L2 normalize into d_out ----
  k_normalize<<<(N_NODES + 7) / 8, 256, 0, stream>>>(S.XW, (float*)d_out, 1024);
}
